// HGTLayer_inter_90228672955105
// MI455X (gfx1250) — matrix operand order repaired and checked by reading, not compiled, not run on hardware
//
#include <hip/hip_runtime.h>
#include <hip/hip_bf16.h>
#include <float.h>

#define N_NODES 16384
#define E_EDGES 65536
#define DGEN    256
#define DRELV   768

typedef __attribute__((ext_vector_type(16))) __bf16 v16bf;
typedef __attribute__((ext_vector_type(8)))  __bf16 v8bf;
typedef __attribute__((ext_vector_type(8)))  float  v8f;

// ---------------------------------------------------------------------------
// Tiled GEMM: C[N x M] = A[N x K] @ W[K x M] + bias, optional epilogue
//   out = C*sigmoid(skip[idx]) + resid*(1-sigmoid(skip[idx]))
// A, W fp32 in global; converted to bf16 while staging through LDS; WMMA bf16
// with fp32 accumulation (v_wmma_f32_16x16x32_bf16).
// Block: 256 threads (8 waves). Tile: 128(rows) x 128(cols) x 32(K).
// Wave w covers rows (w&3)*32, cols (w>>2)*64 => 2x4 16x16 WMMA tiles.
// Software-pipelined: double-buffered LDS, one barrier per K-step; the global
// loads for step i+1 are issued before the WMMAs of step i consume LDS.
// ---------------------------------------------------------------------------
__launch_bounds__(256)
__global__ void gemm_bf16_wmma(const float* __restrict__ A,
                               const float* __restrict__ W,
                               const float* __restrict__ bias,
                               float* __restrict__ Cout,
                               const float* __restrict__ resid,
                               const float* __restrict__ skip, int skipIdx,
                               int K, int M)
{
    __shared__ __bf16 lsA[2][128 * 32];   // [buf][row][k]   2 x 8 KB
    __shared__ __bf16 lsB[2][128 * 32];   // [buf][col][k]   2 x 8 KB (W transposed)

    const int t    = threadIdx.x;
    const int wave = t >> 5;
    const int lane = t & 31;
    const int m0   = blockIdx.y * 128;       // output row base
    const int n0   = blockIdx.x * 128;       // output col base
    const int mW   = (wave & 3) * 32;        // wave row offset in tile
    const int nW   = (wave >> 2) * 64;       // wave col offset in tile
    const int rsel = lane & 15;
    const int hsel = lane >> 4;

    // A staging: 8 threads/row (float4 each), 4 rows per thread
    const int a_rb = t >> 3;                 // 0..31
    const int a_c4 = (t & 7) * 4;            // 0..28
    // W staging: 32 threads/row (float4 each), 4 k-rows per thread
    const int b_kr = t >> 5;                 // 0..7
    const int b_c4 = (t & 31) * 4;           // 0..124

    v8f acc[2][4];
    for (int i = 0; i < 2; ++i)
        for (int j = 0; j < 4; ++j)
            for (int v = 0; v < 8; ++v) acc[i][j][v] = 0.0f;

    float4 ar[4], br[4];
    // prologue: stage tile 0 into registers
    for (int r = 0; r < 4; ++r)
        ar[r] = *(const float4*)(A + (size_t)(m0 + r * 32 + a_rb) * K + a_c4);
    for (int r = 0; r < 4; ++r)
        br[r] = *(const float4*)(W + (size_t)(r * 8 + b_kr) * M + n0 + b_c4);

    const int nsteps = K >> 5;
    for (int s = 0; s < nsteps; ++s) {
        const int cur = s & 1;
        __bf16* bufA = lsA[cur];
        __bf16* bufB = lsB[cur];
        // ---- store staged registers to LDS (fp32 -> bf16) ----
        for (int r = 0; r < 4; ++r) {
            __bf16* p = &bufA[(r * 32 + a_rb) * 32 + a_c4];
            p[0] = (__bf16)ar[r].x; p[1] = (__bf16)ar[r].y;
            p[2] = (__bf16)ar[r].z; p[3] = (__bf16)ar[r].w;
        }
        for (int r = 0; r < 4; ++r) {
            const int kk = r * 8 + b_kr;
            bufB[(b_c4 + 0) * 32 + kk] = (__bf16)br[r].x;
            bufB[(b_c4 + 1) * 32 + kk] = (__bf16)br[r].y;
            bufB[(b_c4 + 2) * 32 + kk] = (__bf16)br[r].z;
            bufB[(b_c4 + 3) * 32 + kk] = (__bf16)br[r].w;
        }
        __syncthreads();
        // ---- issue global loads for the next tile (latency hidden by WMMA) ----
        if (s + 1 < nsteps) {
            const int k0 = (s + 1) * 32;
            for (int r = 0; r < 4; ++r)
                ar[r] = *(const float4*)(A + (size_t)(m0 + r * 32 + a_rb) * K + k0 + a_c4);
            for (int r = 0; r < 4; ++r)
                br[r] = *(const float4*)(W + (size_t)(k0 + r * 8 + b_kr) * M + n0 + b_c4);
        }
        // ---- fragments per ISA 7.12.2 layouts, then 8 WMMAs ----
        union Frag { v16bf v; v8bf h[2]; };
        Frag af[2], bf[4];
        // A (16x32): lane<16 -> K[0..7],K[16..23]; lane>=16 -> K[8..15],K[24..31]
        for (int i = 0; i < 2; ++i) {
            const int row = mW + i * 16 + rsel;
            const int c   = hsel * 8;
            af[i].h[0] = *(const v8bf*)&bufA[row * 32 + c];
            af[i].h[1] = *(const v8bf*)&bufA[row * 32 + 16 + c];
        }
        // B (32x16): lane<16 -> col=lane, K[0..7],K[16..23]; lane>=16 -> K[8..15],K[24..31]
        for (int j = 0; j < 4; ++j) {
            const int col = nW + j * 16 + rsel;
            const int kh  = hsel * 8;
            bf[j].h[0] = *(const v8bf*)&bufB[col * 32 + kh];
            bf[j].h[1] = *(const v8bf*)&bufB[col * 32 + kh + 16];
        }
        for (int i = 0; i < 2; ++i)
            for (int j = 0; j < 4; ++j)
                acc[i][j] = __builtin_amdgcn_wmma_f32_16x16x32_bf16(
                    false, af[i].v, false, bf[j].v,
                    (short)0, acc[i][j], false, false);
        // NOTE: no second barrier needed. The next iteration stores to buffer
        // (s+1)&1, whose last readers (compute of step s-1) are already fenced
        // by this iteration's barrier.
    }

    float alpha = 1.0f, beta = 0.0f;
    if (resid) {
        const float sv = skip[skipIdx];
        alpha = 1.0f / (1.0f + __expf(-sv));
        beta  = 1.0f - alpha;
    }
    // C/D layout: VGPR v -> row = v + 8*(lane>>4), col = lane&15
    for (int i = 0; i < 2; ++i)
        for (int j = 0; j < 4; ++j)
            for (int v = 0; v < 8; ++v) {
                const int row = m0 + mW + i * 16 + hsel * 8 + v;
                const int col = n0 + nW + j * 16 + rsel;
                const size_t idx = (size_t)row * M + col;
                float val = acc[i][j][v] + bias[col];
                if (resid) val = val * alpha + resid[idx] * beta;
                Cout[idx] = val;
            }
}

// ---------------------------------------------------------------------------
// Edge / segment-softmax kernels (one wave32 per edge)
// ---------------------------------------------------------------------------
__device__ inline void atomicMaxFloat(float* addr, float val)
{
    if (val >= 0.0f) atomicMax((int*)addr, __float_as_int(val));
    else             atomicMin((unsigned int*)addr, (unsigned int)__float_as_int(val));
}

__launch_bounds__(256)
__global__ void edge_scores(const int* __restrict__ src, const int* __restrict__ dst,
                            const float* __restrict__ qg, const float* __restrict__ kg,
                            const float* __restrict__ qe, const float* __restrict__ ke,
                            int de, float inv_sqrt_dk,
                            float* __restrict__ att, float* __restrict__ mbuf)
{
    const int edge = blockIdx.x * 8 + (threadIdx.x >> 5);
    const int lane = threadIdx.x & 31;
    const int s = src[edge], d = dst[edge];
    const float* qgp = qg + (size_t)d * DGEN;
    const float* kgp = kg + (size_t)s * DGEN;
    const float* qep = qe + (size_t)d * de;
    const float* kep = ke + (size_t)s * de;
    float acc = 0.0f;
    for (int i = lane; i < DGEN; i += 32) acc += qgp[i] * kgp[i];
    for (int i = lane; i < de;   i += 32) acc += qep[i] * kep[i];
    for (int off = 16; off > 0; off >>= 1) acc += __shfl_xor(acc, off, 32);
    if (lane == 0) {
        const float a = acc * inv_sqrt_dk;
        att[edge] = a;
        atomicMaxFloat(mbuf + d, a);
    }
}

__global__ void edge_expz(const int* __restrict__ dst, float* __restrict__ att,
                          const float* __restrict__ m, float* __restrict__ z)
{
    const int e = blockIdx.x * blockDim.x + threadIdx.x;
    if (e >= E_EDGES) return;
    const int d = dst[e];
    const float ex = expf(att[e] - m[d]);
    att[e] = ex;
    atomicAdd(z + d, ex);
}

__launch_bounds__(256)
__global__ void edge_hacc(const int* __restrict__ src, const int* __restrict__ dst,
                          const float* __restrict__ att, const float* __restrict__ z,
                          const float* __restrict__ vg, const float* __restrict__ ve,
                          int de, float* __restrict__ hg, float* __restrict__ te)
{
    const int edge = blockIdx.x * 8 + (threadIdx.x >> 5);
    const int lane = threadIdx.x & 31;
    const int s = src[edge], d = dst[edge];
    const float zd = z[d];
    const float w  = att[edge] / (zd > 0.0f ? zd : 1.0f);
    const float* vgp = vg + (size_t)s * DGEN;
    const float* vep = ve + (size_t)s * de;
    float* hgp = hg + (size_t)d * DGEN;
    float* tep = te + (size_t)d * de;
    for (int i = lane; i < DGEN; i += 32) atomicAdd(hgp + i, w * vgp[i]);
    for (int i = lane; i < de;   i += 32) atomicAdd(tep + i, w * vep[i]);
}

__global__ void fill_f32(float* __restrict__ p, float v, int n)
{
    const int i = blockIdx.x * blockDim.x + threadIdx.x;
    if (i < n) p[i] = v;
}

__global__ void cnt_update(float* __restrict__ cnt, const float* __restrict__ z)
{
    const int i = blockIdx.x * blockDim.x + threadIdx.x;
    if (i < N_NODES) cnt[i] += (z[i] > 0.0f) ? 1.0f : 0.0f;
}

__global__ void tgen_kernel(float* __restrict__ tg, const float* __restrict__ hg,
                            const float* __restrict__ cnt)
{
    const int i = blockIdx.x * blockDim.x + threadIdx.x;
    if (i < N_NODES * DGEN) tg[i] = hg[i] / fmaxf(cnt[i >> 8], 1.0f);  // DGEN == 256
}

// ---------------------------------------------------------------------------
// Host orchestration
// ---------------------------------------------------------------------------
extern "C" void kernel_launch(void* const* d_in, const int* in_sizes, int n_in,
                              void* d_out, int out_size, void* d_ws, size_t ws_size,
                              hipStream_t stream)
{
    (void)in_sizes; (void)n_in; (void)out_size; (void)ws_size;

    // inputs: x_s/xs_s interleaved per section (insertion order of setup_inputs)
    const float* x[5][2];
    for (int s = 0; s < 5; ++s) {
        x[s][0] = (const float*)d_in[2 * s];
        x[s][1] = (const float*)d_in[2 * s + 1];
    }
    const int* src[4]; const int* dst[4];
    for (int e = 0; e < 4; ++e) {
        src[e] = (const int*)d_in[10 + 2 * e];
        dst[e] = (const int*)d_in[11 + 2 * e];
    }
    // params flattened: per section s: k(W,b), q(W,b), v(W,b), a(W,b); then skip
    const float* Wp[5][4]; const float* Bp[5][4];
    for (int s = 0; s < 5; ++s)
        for (int o = 0; o < 4; ++o) {
            Wp[s][o] = (const float*)d_in[18 + s * 8 + o * 2];
            Bp[s][o] = (const float*)d_in[18 + s * 8 + o * 2 + 1];
        }
    const float* skip = (const float*)d_in[58];

    // workspace carve-up (fp32)
    float* w = (float*)d_ws;
    size_t off = 0;
    auto alloc = [&](size_t n) { float* p = w + off; off += n; return p; };
    float* kg  = alloc((size_t)N_NODES * DGEN);
    float* qg  = alloc((size_t)N_NODES * DGEN);
    float* vg  = alloc((size_t)N_NODES * DGEN);
    float* keb = alloc((size_t)N_NODES * DRELV);
    float* qeb = alloc((size_t)N_NODES * DRELV);
    float* veb = alloc((size_t)N_NODES * DRELV);
    float* hg  = alloc((size_t)N_NODES * DGEN);
    float* te  = alloc((size_t)N_NODES * DRELV);
    float* tg  = alloc((size_t)N_NODES * DGEN);
    float* cnt = alloc(N_NODES);
    float* mb  = alloc(N_NODES);
    float* zb  = alloc(N_NODES);
    float* att = alloc(E_EDGES);

    const int dsec[5] = {DGEN, DGEN, DGEN, DGEN, DRELV};
    const float inv_sqrt_small = 0.044194173824159216f;  // 1/sqrt(512)
    const float inv_sqrt_rel   = 0.03125f;               // 1/sqrt(1024)

    const dim3 blk(256);

    for (int run = 0; run < 2; ++run) {
        float* outBase = (float*)d_out + (size_t)run * N_NODES * (4 * DGEN + DRELV);
        size_t outOff[5];
        outOff[0] = 0;
        for (int s = 1; s < 5; ++s) outOff[s] = outOff[s - 1] + (size_t)N_NODES * dsec[s - 1];

        fill_f32<<<(N_NODES * DGEN) / 256, blk, 0, stream>>>(hg, 0.0f, N_NODES * DGEN);
        fill_f32<<<N_NODES / 256, blk, 0, stream>>>(cnt, 0.0f, N_NODES);

        // k/q/v for 'general'
        {
            float* dstb[3] = {kg, qg, vg};
            dim3 grid(DGEN / 128, N_NODES / 128);
            for (int o = 0; o < 3; ++o)
                gemm_bf16_wmma<<<grid, blk, 0, stream>>>(x[0][run], Wp[0][o], Bp[0][o],
                                                         dstb[o], nullptr, nullptr, 0,
                                                         DGEN, DGEN);
        }

        for (int e = 0; e < 4; ++e) {
            const int s  = e + 1;
            const int de = dsec[s];
            // k/q/v for this edge-type's section
            {
                float* dstb[3] = {keb, qeb, veb};
                dim3 grid(de / 128, N_NODES / 128);
                for (int o = 0; o < 3; ++o)
                    gemm_bf16_wmma<<<grid, blk, 0, stream>>>(x[s][run], Wp[s][o], Bp[s][o],
                                                             dstb[o], nullptr, nullptr, 0,
                                                             de, de);
            }
            fill_f32<<<N_NODES / 256, blk, 0, stream>>>(mb, -FLT_MAX, N_NODES);
            fill_f32<<<N_NODES / 256, blk, 0, stream>>>(zb, 0.0f, N_NODES);
            fill_f32<<<(N_NODES * de) / 256, blk, 0, stream>>>(te, 0.0f, N_NODES * de);

            const float inv_sqrt = (e == 3) ? inv_sqrt_rel : inv_sqrt_small;
            edge_scores<<<E_EDGES / 8, blk, 0, stream>>>(src[e], dst[e], qg, kg, qeb, keb,
                                                         de, inv_sqrt, att, mb);
            edge_expz<<<E_EDGES / 256, blk, 0, stream>>>(dst[e], att, mb, zb);
            cnt_update<<<N_NODES / 256, blk, 0, stream>>>(cnt, zb);
            edge_hacc<<<E_EDGES / 8, blk, 0, stream>>>(src[e], dst[e], att, zb, vg, veb,
                                                       de, hg, te);

            // a-linear for section s, fused sigmoid-skip blend, write to d_out
            dim3 grid(de / 128, N_NODES / 128);
            gemm_bf16_wmma<<<grid, blk, 0, stream>>>(te, Wp[s][3], Bp[s][3],
                                                     outBase + outOff[s], x[s][run],
                                                     skip, s, de, de);
        }

        // t_general = hg_sum / max(cnt,1); then a-linear for 'general'
        tgen_kernel<<<(N_NODES * DGEN) / 256, blk, 0, stream>>>(tg, hg, cnt);
        dim3 grid(DGEN / 128, N_NODES / 128);
        gemm_bf16_wmma<<<grid, blk, 0, stream>>>(tg, Wp[0][3], Bp[0][3],
                                                 outBase + outOff[0], x[0][run],
                                                 skip, 0, DGEN, DGEN);
    }
}
